// PGD_37684043055676
// MI455X (gfx1250) — matrix operand order repaired and checked by reading, not compiled, not run on hardware
//
#include <hip/hip_runtime.h>
#include <hip/hip_bf16.h>

#define DEMB   256
#define BATCH  4
#define SEQ    512
#define NHEAD  8
#define PN     513           // SEQ + 1
#define VOCAB_N 50257
#define VTILE  64
#define WT_ST  72            // padded row stride of transposed W_e tile (bf16 elems)
#define GB_ST  40            // padded column stride of gemm B tile (bf16 elems)

typedef __bf16 bf16_t;
typedef __attribute__((ext_vector_type(16))) __bf16 bf16x16;
typedef __attribute__((ext_vector_type(8)))  float  f32x8;
typedef __attribute__((ext_vector_type(4))) unsigned int u32x4;
typedef __attribute__((ext_vector_type(8))) int          i32x8;
typedef __attribute__((ext_vector_type(4))) int          i32x4;

static __device__ inline f32x8 wmma_bf16(bf16x16 a, bf16x16 b, f32x8 c) {
  return __builtin_amdgcn_wmma_f32_16x16x32_bf16(false, a, false, b, (short)0, c, false, false);
}

static __device__ inline f32x8 zero8() {
  f32x8 z;
#pragma unroll
  for (int i = 0; i < 8; ++i) z[i] = 0.f;
  return z;
}

// A fragment (16x32 bf16) from row-major LDS tile, leading dim ldm.
// lanes 0-15: m=lane, K {0..15}; lanes 16-31: K {16..31}.
// Contiguous 16B chunks -> ds_load_b128 pairs.
static __device__ inline bf16x16 load_A(const bf16_t* s, int row0, int ldm) {
  int lane = threadIdx.x & 31;
  int m    = lane & 15;
  int kh   = (lane >> 4) << 4;
  const bf16_t* p = s + (row0 + m) * ldm;
  bf16x16 a;
#pragma unroll
  for (int i = 0; i < 16; ++i) {
    int k = i + kh;
    a[i] = p[k];
  }
  return a;
}

// B fragment (32x16) where logical B[k][n] = W[n][k]; W row-major, leading dim
// ldw (i.e. W stored n-major / k-contiguous).  lane = n, 16 contiguous K elems
// per lane -> ds_load_b128 pairs.
static __device__ inline bf16x16 load_BT(const bf16_t* w, int col0, int k0, int ldw) {
  int lane = threadIdx.x & 31;
  int n    = lane & 15;
  int kh   = (lane >> 4) << 4;          // 0 or 16
  const bf16_t* p = w + (col0 + n) * ldw + k0 + kh;
  bf16x16 b;
#pragma unroll
  for (int i = 0; i < 16; ++i) b[i] = p[i];
  return b;
}

// ---------------------------------------------------------------------------
// TDM: issue TENSOR_LOAD_TO_LDS of a 2-D bf16 tile (VTILE x DEMB) whose rows
// beyond rows_avail are hardware zero-filled (OOB reads return 0).
// Descriptor packing per CDNA5 ISA.
// This toolchain exposes the 6-arg builtin; the trailing group operands are
// zero (2-D tensor -> descriptor groups 2/3 unused).
// ---------------------------------------------------------------------------
static __device__ inline void tdm_load_tile(unsigned lds_off, const bf16_t* gsrc,
                                            unsigned rows_avail) {
  unsigned long long ga = (unsigned long long)(size_t)gsrc;
  u32x4 g0;
  g0[0] = 1u;                                            // count = 1 valid D#
  g0[1] = lds_off;                                       // lds_addr (bytes)
  g0[2] = (unsigned)(ga & 0xffffffffull);                // global_addr[31:0]
  g0[3] = (unsigned)((ga >> 32) & 0x1ffffffull)          // global_addr[56:32]
        | 0x80000000u;                                   // type = 2 ("image")
  i32x8 g1;
  g1[0] = (int)(1u << 16);                               // data_size = 2 bytes
  g1[1] = (int)((unsigned)DEMB << 16);                   // tensor_dim0 lo16 (=256)
  g1[2] = (int)((rows_avail & 0xffffu) << 16);           // dim0 hi16=0 | tensor_dim1 lo16
  g1[3] = (int)((rows_avail >> 16) |
                ((unsigned)DEMB << 16));                 // tensor_dim1 hi16 | tile_dim0=256
  g1[4] = VTILE;                                         // tile_dim1 = 64 (tile_dim2 = 0)
  g1[5] = DEMB;                                          // tensor_dim0_stride lo32 = 256
  g1[6] = 0;                                             // stride hi | dim1_stride lo
  g1[7] = 0;
  i32x4 gz4 = {0, 0, 0, 0};
  i32x8 gz8 = {0, 0, 0, 0, 0, 0, 0, 0};
  __builtin_amdgcn_tensor_load_to_lds(g0, g1, gz4, gz4, gz8, 0);
}

// ---------------------------------------------------------------------------
// Generic batched WMMA GEMM: C[z] (+)= alpha[z] * A (x) opB(B)
// 64x64 tile / workgroup, 8 waves x 2 accumulators, bf16 operands in LDS
// (B staged k-contiguous for clean ds_load_b128 fragments), fp32 accumulate.
// ---------------------------------------------------------------------------
template <typename TA, typename TB>
__global__ __launch_bounds__(256) void gemm_wmma(
    const TA* __restrict__ A, const TB* __restrict__ B, float* __restrict__ C,
    int M, int N, int Kd, int lda, int ldb, int ldc,
    long long strA, int aDiv, int aMod,
    long long strB, int bDiv, int bMod,
    long long strC, int cDiv,
    int transB, int accum,
    const float* __restrict__ alphaPtr, int alphaMask)
{
  __shared__ __align__(128) bf16_t shA[64 * 32];       // row-major (m, k)
  __shared__ __align__(128) bf16_t shB[64 * GB_ST];    // k-contiguous per column n

  int z = blockIdx.z;
  A += (long long)((z / aDiv) % aMod) * strA;
  B += (long long)((z / bDiv) % bMod) * strB;
  C += (long long)(z / cDiv) * strC;
  float alpha = alphaPtr ? alphaPtr[z & alphaMask] : 1.0f;

  int m0   = blockIdx.y * 64;
  int n0   = blockIdx.x * 64;
  int tid  = threadIdx.x;
  int wave = tid >> 5, lane = tid & 31;
  int mt   = wave & 3;
  int nt0  = (wave >> 2) * 2;

  f32x8 acc0 = zero8(), acc1 = zero8();

  for (int k0 = 0; k0 < Kd; k0 += 32) {
    for (int i = tid; i < 64 * 32; i += 256) {           // stage A (64x32)
      int r = i >> 5, c = i & 31;
      int gm = m0 + r, gk = k0 + c;
      float v = 0.f;
      if (gm < M && gk < Kd) v = (float)A[(long long)gm * lda + gk];
      shA[i] = (bf16_t)v;
    }
    for (int i = tid; i < 32 * 64; i += 256) {           // stage B (k x n=32x64)
      int r = i >> 6, c = i & 63;                        // global-coalesced read
      int gk = k0 + r, gn = n0 + c;
      float v = 0.f;
      if (gn < N && gk < Kd)
        v = transB ? (float)B[(long long)gn * ldb + gk]
                   : (float)B[(long long)gk * ldb + gn];
      shB[c * GB_ST + r] = (bf16_t)v;                    // k-contiguous store
    }
    __syncthreads();
    bf16x16 af = load_A(shA, mt * 16, 32);
    bf16x16 b0 = load_BT(shB, nt0 * 16,      0, GB_ST);
    bf16x16 b1 = load_BT(shB, nt0 * 16 + 16, 0, GB_ST);
    acc0 = wmma_bf16(af, b0, acc0);
    acc1 = wmma_bf16(af, b1, acc1);
    __syncthreads();
  }

  int nl = lane & 15;
  int mh = (lane >> 4) << 3;
#pragma unroll
  for (int j = 0; j < 2; ++j) {
    f32x8 acc = j ? acc1 : acc0;
    int gn = n0 + (nt0 + j) * 16 + nl;
    if (gn >= N) continue;
#pragma unroll
    for (int r = 0; r < 8; ++r) {
      int gm = m0 + mt * 16 + mh + r;
      if (gm >= M) continue;
      float v = acc[r] * alpha;
      long long off = (long long)gm * ldc + gn;
      if (accum) atomicAdd(&C[off], v);
      else       C[off] = v;
    }
  }
}

// ---------------------------------------------------------------------------
// Fused vocab pass (the 420-GFLOP hot loop), flash-softmax style:
//   S = clip(fk @ We_tile^T); P = exp(S); den += rowsum(P); num += P @ We_tile
//   diff = e - num/(den+1e-8); dB[b] += colsum(diff)
// W_e (pre-converted bf16) is streamed by the Tensor Data Mover into a
// double-buffered LDS tile; wave 0 issues TENSOR_LOAD_TO_LDS for tile t+1,
// waits s_wait_tensorcnt<=1 for tile t, then all 8 waves run pure WMMA.
// ---------------------------------------------------------------------------
__global__ __launch_bounds__(256) void vocab_softmax_kernel(
    const bf16_t* __restrict__ fkbf,   // [B*S][256]
    const bf16_t* __restrict__ Webf,   // [VOCAB][256] bf16
    const float*  __restrict__ e,      // [B*S][256]
    float*        __restrict__ diff,   // [B*S][256]
    bf16_t*       __restrict__ diffbf, // [B*S][256]
    float*        __restrict__ dB)     // [B][256]  (pre-zeroed)
{
  __shared__ __align__(128) bf16_t sWe2[2][VTILE * DEMB]; // 64 KB, TDM dest
  __shared__ __align__(128) bf16_t sFk[64 * DEMB];        // 32 KB
  __shared__ __align__(128) bf16_t sWeT[DEMB * WT_ST];    // 36 KB, We tile transposed
  __shared__ __align__(128) bf16_t sE [64 * VTILE];       //  8 KB, exp(S)
  __shared__ float sDen[64];

  int tid  = threadIdx.x;
  int wave = tid >> 5, lane = tid & 31;
  int row0 = blockIdx.x * 64;

  for (int i = tid; i < 64 * DEMB; i += 256)
    sFk[i] = fkbf[(long long)row0 * DEMB + i];
  if (tid < 64) sDen[tid] = 0.f;

  f32x8 accN[4][2];
#pragma unroll
  for (int a = 0; a < 4; ++a) { accN[a][0] = zero8(); accN[a][1] = zero8(); }

  int mt  = wave & 3;
  int nt0 = (wave >> 2) * 2;
  int nl  = lane & 15;
  int mh  = (lane >> 4) << 3;

  const int T = (VOCAB_N + VTILE - 1) / VTILE;            // 786 tiles
  if (wave == 0)
    tdm_load_tile((unsigned)(size_t)&sWe2[0][0], Webf, (unsigned)VOCAB_N);
  __syncthreads();   // covers sFk/sDen init too

  for (int t = 0; t < T; ++t) {
    int cur = t & 1;
    if (wave == 0) {
      if (t + 1 < T) {
        tdm_load_tile((unsigned)(size_t)&sWe2[1 - cur][0],
                      Webf + (long long)(t + 1) * VTILE * DEMB,
                      (unsigned)(VOCAB_N - (t + 1) * VTILE));
        __builtin_amdgcn_s_wait_tensorcnt(1);             // tile t landed
      } else {
        __builtin_amdgcn_s_wait_tensorcnt(0);
      }
    }
    __syncthreads();                                      // tile t visible
    const bf16_t* w = &sWe2[cur][0];
    int v0 = t * VTILE;

    // build transposed copy for phase 2 (contiguous reads, padded writes)
    for (int i = tid; i < VTILE * DEMB; i += 256) {
      int v = i >> 8, d = i & 255;
      sWeT[d * WT_ST + v] = w[i];
    }

    // phase 1: S = fk(64x256) @ We_tile^T(256x64)
    f32x8 s0 = zero8(), s1 = zero8();
#pragma unroll
    for (int kk = 0; kk < DEMB; kk += 32) {
      bf16x16 a  = load_A (sFk + kk, mt * 16, DEMB);
      bf16x16 b0 = load_BT(w, nt0 * 16,      kk, DEMB);
      bf16x16 b1 = load_BT(w, nt0 * 16 + 16, kk, DEMB);
      s0 = wmma_bf16(a, b0, s0);
      s1 = wmma_bf16(a, b1, s1);
    }
#pragma unroll
    for (int j = 0; j < 2; ++j) {
      f32x8 s = j ? s1 : s0;
      int ncol = (nt0 + j) * 16 + nl;
      bool valid = (v0 + ncol) < VOCAB_N;
#pragma unroll
      for (int r = 0; r < 8; ++r) {
        float x = fminf(10.f, fmaxf(-10.f, s[r]));
        float ev = valid ? __expf(x) : 0.f;
        sE[(mt * 16 + mh + r) * VTILE + ncol] = (bf16_t)ev;
      }
    }
    __syncthreads();                                      // sE + sWeT ready

    if (tid < 64) {                                       // denominator partials
      float acc = 0.f;
#pragma unroll 8
      for (int c = 0; c < VTILE; ++c) acc += (float)sE[tid * VTILE + c];
      sDen[tid] += acc;
    }

    // phase 2: num += P(64x64) @ We_tile(64x256); wave -> embed tiles {2w,2w+1}
#pragma unroll
    for (int kk = 0; kk < VTILE; kk += 32) {
      bf16x16 b0 = load_BT(sWeT, (2 * wave) * 16,     kk, WT_ST);
      bf16x16 b1 = load_BT(sWeT, (2 * wave + 1) * 16, kk, WT_ST);
#pragma unroll
      for (int m2 = 0; m2 < 4; ++m2) {
        bf16x16 a = load_A(sE + kk, m2 * 16, VTILE);
        accN[m2][0] = wmma_bf16(a, b0, accN[m2][0]);
        accN[m2][1] = wmma_bf16(a, b1, accN[m2][1]);
      }
    }
    // next iteration's leading barrier separates these reads from overwrites
  }

  __syncthreads();
  int bb = row0 >> 9;
#pragma unroll
  for (int m2 = 0; m2 < 4; ++m2) {
#pragma unroll
    for (int j = 0; j < 2; ++j) {
      int col = (2 * wave + j) * 16 + nl;
      float colsum = 0.f;
#pragma unroll
      for (int r = 0; r < 8; ++r) {
        int lrow = m2 * 16 + mh + r;
        long long grow = row0 + lrow;
        float den = sDen[lrow] + 1e-8f;
        float d = e[grow * DEMB + col] - accN[m2][j][r] / den;
        diff  [grow * DEMB + col] = d;
        diffbf[grow * DEMB + col] = (bf16_t)d;
        colsum += d;
      }
      atomicAdd(&dB[bb * DEMB + col], colsum);
    }
  }
}

// ---------------------------------------------------------------------------
__global__ __launch_bounds__(256) void convert_We_kernel(
    const float* __restrict__ We, bf16_t* __restrict__ Webf)
{
  long long i = (long long)blockIdx.x * 256 + threadIdx.x;  // grid.x == VOCAB_N
  Webf[i] = (bf16_t)We[i];
}

__global__ __launch_bounds__(256) void gather_embed_kernel(
    const int* __restrict__ idx, const float* __restrict__ We,
    float* __restrict__ e)
{
  long long row = blockIdx.x;
  long long tok = idx[row];
  e[row * DEMB + threadIdx.x] = We[tok * DEMB + threadIdx.x];
}

__global__ __launch_bounds__(256) void softmax_rows_kernel(
    const float* __restrict__ scores, bf16_t* __restrict__ Kbf)
{
  __shared__ float red[256];
  long long row = blockIdx.x;
  const float* src = scores + row * PN;
  int tid = threadIdx.x;

  float mx = -3.4e38f;
  for (int i = tid; i < PN; i += 256) mx = fmaxf(mx, src[i]);
  red[tid] = mx; __syncthreads();
  for (int s = 128; s > 0; s >>= 1) {
    if (tid < s) red[tid] = fmaxf(red[tid], red[tid + s]);
    __syncthreads();
  }
  mx = red[0]; __syncthreads();

  float sum = 0.f;
  for (int i = tid; i < PN; i += 256) sum += __expf(src[i] - mx);
  red[tid] = sum; __syncthreads();
  for (int s = 128; s > 0; s >>= 1) {
    if (tid < s) red[tid] += red[tid + s];
    __syncthreads();
  }
  float inv = 1.f / red[0];

  bf16_t* dst = Kbf + row * SEQ;
  for (int i = tid; i < SEQ; i += 256)
    dst[i] = (bf16_t)(__expf(src[i] - mx) * inv);
}

__global__ __launch_bounds__(256) void update_fk_kernel(
    float* __restrict__ fk, bf16_t* __restrict__ fkbf,
    const float* __restrict__ dA, const float* __restrict__ dBraw,
    const float* __restrict__ B_LR)
{
  int b = blockIdx.y;
  int n = blockIdx.x;                   // 0..512
  int d = threadIdx.x;
  long long o = ((long long)b * PN + n) * DEMB + d;
  float v = fk[o] + (dA[o] + dBraw[b * DEMB + d] * B_LR[0]) * (1.0f / (float)SEQ);
  fk[o] = v;
  if (n < SEQ) fkbf[((long long)b * SEQ + n) * DEMB + d] = (bf16_t)v;
}

// ---------------------------------------------------------------------------
extern "C" void kernel_launch(void* const* d_in, const int* in_sizes, int n_in,
                              void* d_out, int out_size, void* d_ws, size_t ws_size,
                              hipStream_t stream)
{
  (void)in_sizes; (void)n_in; (void)out_size; (void)ws_size;

  const int*   idx  = (const int*)  d_in[0];
  const float* W_e  = (const float*)d_in[1];
  const float* W_p  = (const float*)d_in[2];
  const float* W_k  = (const float*)d_in[3];
  const float* W_q  = (const float*)d_in[4];
  const float* W_v  = (const float*)d_in[5];
  const float* A_LR = (const float*)d_in[6];
  const float* B_LR = (const float*)d_in[7];
  float* logits = (float*)d_out;

  char* ws = (char*)d_ws;
  auto take = [&](size_t bytes) -> char* {
    char* p = ws;
    ws += (bytes + 255) & ~(size_t)255;
    return p;
  };

  const int ROWS = BATCH * SEQ;                                     // 2048
  float*  e_emb = (float*) take((size_t)ROWS * DEMB * 4);           // 2 MB
  bf16_t* Webf  = (bf16_t*)take((size_t)VOCAB_N * DEMB * 2);        // 25.7 MB
  bf16_t* Kbf   = (bf16_t*)take((size_t)NHEAD * PN * SEQ * 2);      // 4.2 MB
  float*  fk    = (float*) take((size_t)BATCH * PN * DEMB * 4);     // 2.1 MB
  bf16_t* fkbf  = (bf16_t*)take((size_t)ROWS * DEMB * 2);           // 1 MB
  char*   scratch = take((size_t)24 * 1024 * 1024);                 // shared

  float* x_i  = (float*)scratch;
  float* x_j  = x_i + (size_t)NHEAD * PN * DEMB;
  float* scor = x_j + (size_t)NHEAD * PN * DEMB;
  float*  diff   = (float*)scratch;
  bf16_t* diffbf = (bf16_t*)(diff + (size_t)ROWS * DEMB);
  float*  Vbuf   = (float*)(diffbf + (size_t)ROWS * DEMB);
  float*  dA     = Vbuf + (size_t)BATCH * NHEAD * SEQ * DEMB;
  float*  dB     = dA + (size_t)BATCH * PN * DEMB;

  const int BIG = 1 << 30;
  dim3 blk(256);

  (void)hipMemsetAsync(fk,   0, (size_t)BATCH * PN * DEMB * 4, stream);
  (void)hipMemsetAsync(fkbf, 0, (size_t)ROWS * DEMB * 2, stream);

  convert_We_kernel<<<dim3(VOCAB_N), blk, 0, stream>>>(W_e, Webf);
  gather_embed_kernel<<<dim3(ROWS), blk, 0, stream>>>(idx, W_e, e_emb);

  {
    dim3 g(DEMB / 64, (PN + 63) / 64, NHEAD);
    gemm_wmma<float, float><<<g, blk, 0, stream>>>(
        W_p, W_k, x_i, PN, DEMB, DEMB, DEMB, DEMB, DEMB,
        0LL, 1, BIG, (long long)DEMB * DEMB, 1, BIG, (long long)PN * DEMB, 1,
        0, 0, nullptr, 0);
    gemm_wmma<float, float><<<g, blk, 0, stream>>>(
        W_p, W_q, x_j, PN, DEMB, DEMB, DEMB, DEMB, DEMB,
        0LL, 1, BIG, (long long)DEMB * DEMB, 1, BIG, (long long)PN * DEMB, 1,
        0, 0, nullptr, 0);
  }
  {
    dim3 g((PN + 63) / 64, (PN + 63) / 64, NHEAD);
    gemm_wmma<float, float><<<g, blk, 0, stream>>>(
        x_j, x_i, scor, PN, PN, DEMB, DEMB, DEMB, PN,
        (long long)PN * DEMB, 1, BIG, (long long)PN * DEMB, 1, BIG,
        (long long)PN * PN, 1, 1, 0, nullptr, 0);
  }
  softmax_rows_kernel<<<dim3(NHEAD * PN), blk, 0, stream>>>(scor, Kbf);

  for (int layer = 0; layer < 4; ++layer) {
    (void)hipMemsetAsync(dA, 0, (size_t)BATCH * PN * DEMB * 4, stream);
    (void)hipMemsetAsync(dB, 0, (size_t)BATCH * DEMB * 4, stream);

    vocab_softmax_kernel<<<dim3(ROWS / 64), blk, 0, stream>>>(
        fkbf, Webf, e_emb, diff, diffbf, dB);

    {   // V[b,h] = diff_b @ W_v[h]          (z = b*8 + h)
      dim3 g(DEMB / 64, SEQ / 64, BATCH * NHEAD);
      gemm_wmma<bf16_t, float><<<g, blk, 0, stream>>>(
          diffbf, W_v, Vbuf, SEQ, DEMB, DEMB, DEMB, DEMB, DEMB,
          (long long)SEQ * DEMB, NHEAD, BIG,
          (long long)DEMB * DEMB, 1, NHEAD,
          (long long)SEQ * DEMB, 1,
          0, 0, nullptr, 0);
    }
    {   // dA[b] += A_LR[h] * K[h] @ V[b,h]  (z = b*8 + h, atomic accumulate)
      dim3 g(DEMB / 64, (PN + 63) / 64, BATCH * NHEAD);
      gemm_wmma<bf16_t, float><<<g, blk, 0, stream>>>(
          Kbf, Vbuf, dA, PN, DEMB, SEQ, SEQ, DEMB, DEMB,
          (long long)PN * SEQ, 1, NHEAD,
          (long long)SEQ * DEMB, 1, BIG,
          (long long)PN * DEMB, NHEAD,
          0, 1, A_LR, NHEAD - 1);
    }
    update_fk_kernel<<<dim3(PN, BATCH), blk, 0, stream>>>(fk, fkbf, dA, dB, B_LR);
  }

  {   // logits = f_k[:, -1] @ W_e^T
    dim3 g((VOCAB_N + 63) / 64, 1, 1);
    gemm_wmma<float, float><<<g, blk, 0, stream>>>(
        fk + (size_t)SEQ * DEMB, W_e, logits, BATCH, VOCAB_N, DEMB,
        PN * DEMB, DEMB, VOCAB_N,
        0LL, 1, BIG, 0LL, 1, BIG, 0LL, 1,
        1, 0, nullptr, 0);
  }
}
